// ResidualBlock_10058813407580
// MI455X (gfx1250) — matrix operand order repaired and checked by reading, not compiled, not run on hardware
//
#include <hip/hip_runtime.h>
#include <hip/hip_bf16.h>

// ---------------------------------------------------------------------------
// ResidualBlock with per-sample (hypernetwork) 3x3 convs, train-mode BN, ReLU.
// Conv as implicit GEMM on the CDNA5 WMMA path:
//   per sample: M=64 (out ch), N=12544 (112*112), K=576, permuted K = r*64+i
//   v_wmma_f32_16x16x32_bf16, f32 accumulate.
// Activations are pre-padded (zero border) and pre-converted to bf16 (with
// BN+ReLU fused for conv2's input), so the conv hot loop is pure
// literal-offset d16 gathers + b128 weight loads + WMMA.
// ---------------------------------------------------------------------------

typedef __attribute__((ext_vector_type(16))) __bf16 v16bf;
typedef __attribute__((ext_vector_type(8)))  float  v8f;

#define B_    32
#define C_    64
#define HW_   12544       // 112*112
#define HWDIM 112
#define PDIM  114         // padded spatial dim
#define PHW_  12996       // 114*114
#define KDIM  576         // 64 * 9
#define KSTEPS 18         // 576 / 32
#define PAIRS_PER_SAMPLE 392                     // (12544/16)/2 N-tile pairs
#define TOTAL_PAIRS (B_ * PAIRS_PER_SAMPLE)      // 12544
#define WAVES_PER_BLOCK 8
#define CONV_GRID (TOTAL_PAIRS / WAVES_PER_BLOCK)   // 1568
#define WPK_TASKS (B_ * KSTEPS * 4 * 32)         // 73728 (b,kt,mt,lane)
#define WPK_ELEMS ((size_t)WPK_TASKS * 16)       // bf16 elements per packed conv
#define PAD_ELEMS ((size_t)B_ * C_ * PHW_)       // 26,615,808 bf16
#define NSTAT 401408.0f   // 32*112*112, BN reduction count per channel

// K-index inside a 32-wide K-step for the 16-bit WMMA *A* register layout.
// contiguous: lanes 0-15 -> K=e ; lanes 16-31 -> K=16+e (matches B gather).
static __device__ __forceinline__ int a_klocal(int e, int half) {
    return 16 * half + e;
}

// ---------------------------------------------------------------------------
// Pre-pack per-sample weights [64 x (64x3x3)] f32 into bf16 in the WMMA
// A-register layout with PERMUTED K: k' = r*64 + i  (r = 3x3 pos, i = in ch).
// wpk[((b*18+kt)*4+mt)*32+lane][e], one thread per (b,kt,mt,lane).
// ---------------------------------------------------------------------------
__global__ __launch_bounds__(256) void repack_w(
    const float* __restrict__ w, unsigned short* __restrict__ wpk)
{
    const int t    = blockIdx.x * 256 + threadIdx.x;  // [0, 73728)
    const int lane = t & 31;
    const int mt   = (t >> 5) & 3;
    const int bkt  = t >> 7;             // b*18 + kt
    const int b    = bkt / KSTEPS;
    const int kt   = bkt - b * KSTEPS;
    const int half = lane >> 4;
    const int lan  = lane & 15;

    const float* __restrict__ wrow =
        w + (size_t)b * (C_ * KDIM) + (size_t)(mt * 16 + lan) * KDIM;
    v16bf v;
#pragma unroll
    for (int e = 0; e < 16; ++e) {
        const int kp = kt * 32 + a_klocal(e, half);  // permuted K index
        const int r  = kp >> 6;                      // 3x3 position
        const int i  = kp & 63;                      // input channel
        v[e] = (__bf16)wrow[i * 9 + r];
    }
    *(v16bf*)(wpk + (size_t)t * 16) = v;
}

// ---------------------------------------------------------------------------
// Pad to [B,64,114,114] bf16 with zero border; optionally fuse BN+ReLU
// (conv2 input path). One thread per padded element.
// ---------------------------------------------------------------------------
template <bool BN>
__global__ __launch_bounds__(256) void pad_bf16(
    const float* __restrict__ src,    // [B, 64, 112, 112]
    const float* __restrict__ scale,
    const float* __restrict__ bias,
    __bf16* __restrict__ dst)         // [B, 64, 114, 114]
{
    const size_t t  = (size_t)blockIdx.x * 256 + threadIdx.x;  // [0, PAD_ELEMS)
    const size_t bc = t / PHW_;
    const int   pp  = (int)(t - bc * PHW_);
    const int   hh  = pp / PDIM;
    const int   ww  = pp - hh * PDIM;
    float v = 0.0f;
    if (hh >= 1 && hh <= HWDIM && ww >= 1 && ww <= HWDIM) {
        v = src[bc * HW_ + (size_t)(hh - 1) * HWDIM + (ww - 1)];
        if (BN) {
            const int c = (int)(bc & (C_ - 1));
            v = fmaxf(v * scale[c] + bias[c], 0.0f);
        }
    }
    dst[t] = (__bf16)v;
}

// ---------------------------------------------------------------------------
// One wave: a pair of N-tiles (32 pixels) x all 64 output channels (4 M-tiles).
// Fully unrolled 9 (3x3 pos) x 2 (channel-halves) K-steps. Per K-step:
// 32 literal-offset d16 gathers (B), 4 b128 loads (A), 8 back-to-back WMMAs.
// ---------------------------------------------------------------------------
__global__ __launch_bounds__(256) void conv_wmma(
    const __bf16* __restrict__ xpad,          // [B, 64, 114, 114] bf16
    const unsigned short* __restrict__ wpk,   // packed bf16 weights
    float* __restrict__ out)                  // [B, 64, 112, 112] raw conv out
{
    const int lane = threadIdx.x & 31;
    const int wave = threadIdx.x >> 5;
    const int t    = blockIdx.x * WAVES_PER_BLOCK + wave;  // pair id, uniform per wave
    const int b    = t / PAIRS_PER_SAMPLE;
    const int n0   = (t - b * PAIRS_PER_SAMPLE) * 32;
    const int half = lane >> 4;
    const int lan  = lane & 15;

    const v16bf* __restrict__ wa =
        (const v16bf*)wpk + (size_t)b * (KSTEPS * 4 * 32) + lane;

    const int nA = n0 + lan;         // pixel for N-tile 0
    const int nB = nA + 16;          // pixel for N-tile 1
    const int hA = nA / HWDIM, wcA = nA - hA * HWDIM;
    const int hB = nB / HWDIM, wcB = nB - hB * HWDIM;

    // per-lane base pointers into the padded bf16 input (half -> channels 16..31)
    const __bf16* __restrict__ xh =
        xpad + (size_t)b * (C_ * PHW_) + (size_t)(half * 16) * PHW_;
    const __bf16* __restrict__ pA = xh + hA * PDIM + wcA;
    const __bf16* __restrict__ pB = xh + hB * PDIM + wcB;

    v8f acc[4][2] = {};

#pragma unroll
    for (int r9 = 0; r9 < 9; ++r9) {
        const int kh = r9 / 3;
        const int kw = r9 - kh * 3;
        const int sp = kh * PDIM + kw;      // literal spatial offset (padded coords)
#pragma unroll
        for (int c32 = 0; c32 < 2; ++c32) {
            const int kt = r9 * 2 + c32;
            // B tiles: channel i = half*16 + c32*32 + 2v + (e&1); all offsets literal
            v16bf bt0, bt1;
#pragma unroll
            for (int e = 0; e < 16; ++e) {
                const int ip  = c32 * 32 + 2 * (e >> 1) + (e & 1);
                const int off = sp + ip * PHW_;
                bt0[e] = pA[off];
                bt1[e] = pB[off];
            }
#pragma unroll
            for (int mt = 0; mt < 4; ++mt) {
                const v16bf at = wa[(size_t)(kt * 4 + mt) * 32];
                acc[mt][0] = __builtin_amdgcn_wmma_f32_16x16x32_bf16(
                    false, at, false, bt0, (short)0, acc[mt][0], false, false);
                acc[mt][1] = __builtin_amdgcn_wmma_f32_16x16x32_bf16(
                    false, at, false, bt1, (short)0, acc[mt][1], false, false);
            }
        }
    }

    // ---- store raw conv output (C/D layout: VGPR r -> M = mt*16 + 8*half + r)
#pragma unroll
    for (int mt = 0; mt < 4; ++mt) {
#pragma unroll
        for (int r = 0; r < 8; ++r) {
            const int c = mt * 16 + half * 8 + r;
            float* __restrict__ po = out + ((size_t)b * C_ + c) * HW_;
            po[nA] = acc[mt][0][r];
            po[nB] = acc[mt][1][r];
        }
    }
}

// Deterministic per-channel batch stats (mean/var over N,H,W) -> fused BN
// coefficients: scale = gamma * rstd, bias = beta - mean * gamma * rstd.
__global__ __launch_bounds__(256) void reduce_bn(
    const float* __restrict__ act,    // [B, 64, 112, 112]
    const float* __restrict__ gamma,
    const float* __restrict__ beta,
    float* __restrict__ scale,
    float* __restrict__ bias)
{
    const int c   = blockIdx.x;       // 64 blocks, one channel each
    const int tid = threadIdx.x;
    float s = 0.0f, q = 0.0f;
    for (int b = 0; b < B_; ++b) {
        const float* __restrict__ plane = act + ((size_t)b * C_ + c) * HW_;
        for (int n4 = tid * 4; n4 < HW_; n4 += 256 * 4) {
            const float4 v = *(const float4*)(plane + n4);
            s += v.x + v.y + v.z + v.w;
            q += v.x * v.x + v.y * v.y + v.z * v.z + v.w * v.w;
        }
    }
    __shared__ float ss[256];
    __shared__ float sq[256];
    ss[tid] = s; sq[tid] = q;
    __syncthreads();
    for (int o = 128; o > 0; o >>= 1) {
        if (tid < o) { ss[tid] += ss[tid + o]; sq[tid] += sq[tid + o]; }
        __syncthreads();
    }
    if (tid == 0) {
        const float invN = 1.0f / NSTAT;
        const float m    = ss[0] * invN;
        const float var  = sq[0] * invN - m * m;
        const float rstd = rsqrtf(var + 1e-5f);
        const float g    = gamma[c] * rstd;
        scale[c] = g;
        bias[c]  = beta[c] - m * g;
    }
}

// Epilogue: out = relu(bn2(conv2_raw) + residual), in place on d_out.
__global__ __launch_bounds__(256) void bn_res_relu(
    const float* __restrict__ x,
    const float* __restrict__ scale,
    const float* __restrict__ bias,
    float* __restrict__ out)
{
    const size_t idx = ((size_t)blockIdx.x * blockDim.x + threadIdx.x) * 4;
    const int c = (int)((idx / HW_) % C_);   // HW_ % 4 == 0 -> c uniform in the 4
    const float s = scale[c];
    const float bb = bias[c];
    const float4 o  = *(const float4*)(out + idx);
    const float4 xv = *(const float4*)(x + idx);
    float4 r;
    r.x = fmaxf(o.x * s + bb + xv.x, 0.0f);
    r.y = fmaxf(o.y * s + bb + xv.y, 0.0f);
    r.z = fmaxf(o.z * s + bb + xv.z, 0.0f);
    r.w = fmaxf(o.w * s + bb + xv.w, 0.0f);
    *(float4*)(out + idx) = r;
}

extern "C" void kernel_launch(void* const* d_in, const int* in_sizes, int n_in,
                              void* d_out, int out_size, void* d_ws, size_t ws_size,
                              hipStream_t stream) {
    const float* x      = (const float*)d_in[0];
    const float* w1     = (const float*)d_in[1];
    const float* w2     = (const float*)d_in[2];
    const float* gamma1 = (const float*)d_in[3];
    const float* beta1  = (const float*)d_in[4];
    const float* gamma2 = (const float*)d_in[5];
    const float* beta2  = (const float*)d_in[6];
    float* out = (float*)d_out;

    const size_t NELEM = (size_t)B_ * C_ * HW_;   // 25,690,112 floats

    unsigned short* wpk1 = (unsigned short*)d_ws;     // 2.36 MB packed bf16
    unsigned short* wpk2 = wpk1 + WPK_ELEMS;          // 2.36 MB packed bf16
    __bf16* xpad  = (__bf16*)(wpk2 + WPK_ELEMS);      // 53.2 MB, reused by both convs
    float*  act1  = (float*)(xpad + PAD_ELEMS);       // conv1 raw output, f32
    float*  stats = act1 + NELEM;
    float*  scale1 = stats;
    float*  bias1  = stats + 64;
    float*  scale2 = stats + 128;
    float*  bias2  = stats + 192;

    const int PAD_GRID = (int)(PAD_ELEMS / 256);      // exact: 103,968

    // 0) pack both per-sample weight sets into WMMA-native bf16 layout
    repack_w<<<WPK_TASKS / 256, 256, 0, stream>>>(w1, wpk1);
    repack_w<<<WPK_TASKS / 256, 256, 0, stream>>>(w2, wpk2);
    // 1) pad x -> bf16 [B,64,114,114]
    pad_bf16<false><<<PAD_GRID, 256, 0, stream>>>(x, nullptr, nullptr, xpad);
    // 2) conv1 raw -> act1 (f32, for exact batch stats)
    conv_wmma<<<CONV_GRID, 256, 0, stream>>>(xpad, wpk1, act1);
    // 3) BN1 batch stats -> fused scale1/bias1
    reduce_bn<<<C_, 256, 0, stream>>>(act1, gamma1, beta1, scale1, bias1);
    // 4) pad BN1(act1)+ReLU -> bf16, overwriting xpad
    pad_bf16<true><<<PAD_GRID, 256, 0, stream>>>(act1, scale1, bias1, xpad);
    // 5) conv2 -> d_out (raw)
    conv_wmma<<<CONV_GRID, 256, 0, stream>>>(xpad, wpk2, out);
    // 6) BN2 batch stats -> fused scale2/bias2
    reduce_bn<<<C_, 256, 0, stream>>>(out, gamma2, beta2, scale2, bias2);
    // 7) out = relu(bn2(out) + x), in place, float4
    bn_res_relu<<<(int)(NELEM / 4 / 256), 256, 0, stream>>>(x, scale2, bias2, out);
}
